// Model_5136780886035
// MI455X (gfx1250) — matrix operand order repaired and checked by reading, not compiled, not run on hardware
//
#include <hip/hip_runtime.h>
#include <math.h>

#define HIDDEN 128
#define NLAYERS 6

typedef __attribute__((ext_vector_type(16))) _Float16 v16h;
typedef __attribute__((ext_vector_type(8)))  _Float16 v8h;
typedef __attribute__((ext_vector_type(8)))  float    v8f;

union AFrag { v16h v; v8h h[2]; };

// ---- CDNA5 async global->LDS path (ASYNCcnt), guarded for toolchain safety ----
#if defined(__AMDGCN__) && __has_builtin(__builtin_amdgcn_global_load_async_to_lds_b128) && \
    __has_builtin(__builtin_amdgcn_s_wait_asynccnt)
#define HAVE_ASYNC_LDS 1
typedef __attribute__((__vector_size__(4 * sizeof(int)))) int v4i_t;
__device__ __forceinline__ void async_copy16(const void* g, void* l) {
    __builtin_amdgcn_global_load_async_to_lds_b128(
        (__attribute__((address_space(1))) v4i_t*)g,
        (__attribute__((address_space(3))) v4i_t*)l, 0, 0);
}
__device__ __forceinline__ void async_wait0() { __builtin_amdgcn_s_wait_asynccnt(0); }
#else
#define HAVE_ASYNC_LDS 0
__device__ __forceinline__ void async_copy16(const void* g, void* l) {
    *(uint4*)l = *(const uint4*)g;
}
__device__ __forceinline__ void async_wait0() {}
#endif

// ---------------------------------------------------------------------------
// Weight transpose + f32->f16 convert:  src[l][k][n] -> dst[l][n][k]
// ---------------------------------------------------------------------------
__global__ void convt_kernel(const float* __restrict__ src, _Float16* __restrict__ dst,
                             int K, int Nn, int total) {
    int i = blockIdx.x * blockDim.x + threadIdx.x;
    if (i >= total) return;
    int kn = K * Nn;
    int l = i / kn;
    int r = i - l * kn;
    int k = r / Nn;
    int n = r - k * Nn;
    dst[(size_t)l * kn + (size_t)n * K + k] = (_Float16)src[i];
}

__global__ void zero_kernel(float* __restrict__ p, size_t n) {
    size_t i = (size_t)blockIdx.x * blockDim.x + threadIdx.x;
    if (i < n) p[i] = 0.0f;
}

// degree count (scatter of ones by dst)
__global__ void count_kernel(const int* __restrict__ edge_index, float* __restrict__ cnt, int E) {
    int e = blockIdx.x * blockDim.x + threadIdx.x;
    if (e < E) atomicAdd(&cnt[edge_index[E + e]], 1.0f);
}

// encoder: ef[e][n] = sum_k edge_attr[e][k] * enc_w[k][n] + enc_b[n]   (K=3), ef stored f16
__global__ void encoder_kernel(const float* __restrict__ ea, const float* __restrict__ enc_w,
                               const float* __restrict__ enc_b, _Float16* __restrict__ ef, int E) {
    int g = blockIdx.x * blockDim.x + threadIdx.x;
    if (g >= E * HIDDEN) return;
    int e = g >> 7;
    int n = g & (HIDDEN - 1);
    float a0 = ea[3 * e + 0], a1 = ea[3 * e + 1], a2 = ea[3 * e + 2];
    ef[g] = (_Float16)(a0 * enc_w[n] + a1 * enc_w[HIDDEN + n] + a2 * enc_w[2 * HIDDEN + n] + enc_b[n]);
}

// ---------------------------------------------------------------------------
// Fused edge MLP:  m = relu([x[dst]|x[src]|ef] @ W1 + b1) @ W2 + b2
//                  ef += m ;  sums[dst] += m    (64 edges / block, 4 waves)
// ---------------------------------------------------------------------------
__global__ __launch_bounds__(128)
void edge_mlp_kernel(const float* __restrict__ x,
                     _Float16* __restrict__ ef,
                     float* __restrict__ sums,
                     const int* __restrict__ edge_index,
                     const _Float16* __restrict__ w1t,   // [128][384]  (n-major)
                     const _Float16* __restrict__ w2t,   // [128][128]
                     const float* __restrict__ b1,
                     const float* __restrict__ b2,
                     int E) {
    __shared__ alignas(32) _Float16 As[64 * 32];     //  4 KB : A chunk  (64 rows x K32)
    __shared__ alignas(32) _Float16 Bs[128 * 32];    //  8 KB : W1^T chunk
    __shared__ alignas(32) _Float16 Hs[64 * 128];    // 16 KB : relu hidden
    __shared__ alignas(32) _Float16 W2s[128 * 128];  // 32 KB : full W2^T
    __shared__ int sidx[2][64];                      // src / dst

    const int tid   = threadIdx.x;
    const int lane  = tid & 31;
    const int wave  = tid >> 5;
    const int mrow  = lane & 15;        // M (A) or N (B/D) within a 16-tile
    const int khalf = lane >> 4;        // which K-half this lane owns
    const int e0    = blockIdx.x * 64;

    if (tid < 64) {
        int e = e0 + tid; if (e >= E) e = E - 1;
        sidx[0][tid] = edge_index[e];        // src
        sidx[1][tid] = edge_index[E + e];    // dst
    }
    // stage whole W2^T via async DMA (2048 x 16B segments)
    for (int i = tid; i < 2048; i += 128)
        async_copy16((const uint4*)w2t + i, (uint4*)W2s + i);

    v8f c1[8];
    #pragma unroll
    for (int i = 0; i < 8; ++i) c1[i] = {};

    // ---------------- GEMM1: K = 384 in 12 chunks of 32 ----------------
    for (int kc = 0; kc < 12; ++kc) {
        __syncthreads();                       // frag reads of previous chunk done
        if (kc < 8) {
            // gather + f32->f16 convert from x (2 threads per row, 16 cols each)
            int r  = tid >> 1;
            int cs = (tid & 1) * 16;
            int j0 = kc * 32 + cs;             // column of the virtual concat
            const float* srow = (j0 < 128)
                ? x + (size_t)sidx[1][r] * HIDDEN + j0
                : x + (size_t)sidx[0][r] * HIDDEN + (j0 - 128);
            _Float16* dp = &As[r * 32 + cs];
            #pragma unroll
            for (int q = 0; q < 16; ++q) dp[q] = (_Float16)srow[q];
        } else {
            // ef already f16: raw async copy, 64 rows x 4 segs of 16B
            for (int i = tid; i < 256; i += 128) {
                int r = i >> 2, s = i & 3;
                async_copy16(ef + (size_t)(e0 + r) * HIDDEN + (kc - 8) * 32 + s * 8,
                             &As[r * 32 + s * 8]);
            }
        }
        // stage W1^T chunk via async DMA: seg i -> n = i>>2, s = i&3 (8 halves each)
        for (int i = tid; i < 512; i += 128) {
            int n = i >> 2, s = i & 3;
            const _Float16* gp = w1t + (size_t)n * 384 + kc * 32 + s * 8;
            async_copy16(gp, &Bs[n * 32 + s * 8]);
            if (kc < 11) __builtin_prefetch(gp + 32, 0, 0);   // next chunk
        }
        async_wait0();
        __syncthreads();

        AFrag a;
        a.h[0] = *(const v8h*)&As[(wave * 16 + mrow) * 32 + khalf * 16];
        a.h[1] = *(const v8h*)&As[(wave * 16 + mrow) * 32 + khalf * 16 + 8];
        #pragma unroll
        for (int nt = 0; nt < 8; ++nt) {
            v16h b = *(const v16h*)&Bs[(nt * 16 + mrow) * 32 + khalf * 16];
            c1[nt] = __builtin_amdgcn_wmma_f32_16x16x32_f16(
                false, a.v, false, b, (short)0, c1[nt], false, false);
        }
    }

    // bias + relu -> Hs (f16)
    #pragma unroll
    for (int nt = 0; nt < 8; ++nt) {
        int col = nt * 16 + mrow;
        float bias = b1[col];
        #pragma unroll
        for (int r = 0; r < 8; ++r) {
            int row = wave * 16 + r + khalf * 8;
            float v = c1[nt][r] + bias;
            Hs[row * 128 + col] = (_Float16)(v > 0.0f ? v : 0.0f);
        }
    }

    // ---------------- GEMM2: K = 128 in 4 chunks of 32 ----------------
    v8f c2[8];
    #pragma unroll
    for (int i = 0; i < 8; ++i) c2[i] = {};
    #pragma unroll
    for (int kc = 0; kc < 4; ++kc) {
        int kb = kc * 32;
        AFrag a;
        a.h[0] = *(const v8h*)&Hs[(wave * 16 + mrow) * 128 + kb + khalf * 16];
        a.h[1] = *(const v8h*)&Hs[(wave * 16 + mrow) * 128 + kb + khalf * 16 + 8];
        #pragma unroll
        for (int nt = 0; nt < 8; ++nt) {
            v16h b = *(const v16h*)&W2s[(nt * 16 + mrow) * 128 + kb + khalf * 16];
            c2[nt] = __builtin_amdgcn_wmma_f32_16x16x32_f16(
                false, a.v, false, b, (short)0, c2[nt], false, false);
        }
    }

    // epilogue: m = c2 + b2 ;  ef += m (f16 RMW) ;  sums[dst] += m (f32 atomics)
    #pragma unroll
    for (int nt = 0; nt < 8; ++nt) {
        int col = nt * 16 + mrow;
        float bias = b2[col];
        #pragma unroll
        for (int r = 0; r < 8; ++r) {
            int erow = wave * 16 + r + khalf * 8;
            int e = e0 + erow;
            if (e < E) {
                float mv = c2[nt][r] + bias;
                atomicAdd(&sums[(size_t)sidx[1][erow] * HIDDEN + col], mv);
                _Float16* efp = ef + (size_t)e * HIDDEN + col;
                *efp = (_Float16)((float)*efp + mv);
            }
        }
    }
}

// ---------------------------------------------------------------------------
// Fused node MLP: x += relu([x | sums/cnt] @ W1 + b1) @ W2 + b2
// ---------------------------------------------------------------------------
__global__ __launch_bounds__(128)
void node_mlp_kernel(float* __restrict__ x,
                     const float* __restrict__ sums,
                     const float* __restrict__ cnt,
                     const _Float16* __restrict__ w1t,   // [128][256]
                     const _Float16* __restrict__ w2t,   // [128][128]
                     const float* __restrict__ b1,
                     const float* __restrict__ b2,
                     int N) {
    __shared__ alignas(32) _Float16 As[64 * 32];
    __shared__ alignas(32) _Float16 Bs[128 * 32];
    __shared__ alignas(32) _Float16 Hs[64 * 128];
    __shared__ alignas(32) _Float16 W2s[128 * 128];

    const int tid   = threadIdx.x;
    const int lane  = tid & 31;
    const int wave  = tid >> 5;
    const int mrow  = lane & 15;
    const int khalf = lane >> 4;
    const int n0    = blockIdx.x * 64;

    for (int i = tid; i < 2048; i += 128)
        async_copy16((const uint4*)w2t + i, (uint4*)W2s + i);

    v8f c1[8];
    #pragma unroll
    for (int i = 0; i < 8; ++i) c1[i] = {};

    // GEMM1: K = 256 in 8 chunks
    for (int kc = 0; kc < 8; ++kc) {
        __syncthreads();
        {
            int r   = tid >> 1;
            int row = n0 + r;
            int cs  = (tid & 1) * 16;
            int j0  = kc * 32 + cs;
            _Float16* dp = &As[r * 32 + cs];
            if (row < N) {
                if (j0 < 128) {
                    const float* s = x + (size_t)row * HIDDEN + j0;
                    #pragma unroll
                    for (int q = 0; q < 16; ++q) dp[q] = (_Float16)s[q];
                } else {
                    float inv = 1.0f / fmaxf(cnt[row], 1.0f);
                    const float* s = sums + (size_t)row * HIDDEN + (j0 - 128);
                    #pragma unroll
                    for (int q = 0; q < 16; ++q) dp[q] = (_Float16)(s[q] * inv);
                }
            } else {
                #pragma unroll
                for (int q = 0; q < 16; ++q) dp[q] = (_Float16)0.0f;
            }
        }
        for (int i = tid; i < 512; i += 128) {
            int n = i >> 2, s = i & 3;
            const _Float16* gp = w1t + (size_t)n * 256 + kc * 32 + s * 8;
            async_copy16(gp, &Bs[n * 32 + s * 8]);
            if (kc < 7) __builtin_prefetch(gp + 32, 0, 0);
        }
        async_wait0();
        __syncthreads();

        AFrag a;
        a.h[0] = *(const v8h*)&As[(wave * 16 + mrow) * 32 + khalf * 16];
        a.h[1] = *(const v8h*)&As[(wave * 16 + mrow) * 32 + khalf * 16 + 8];
        #pragma unroll
        for (int nt = 0; nt < 8; ++nt) {
            v16h b = *(const v16h*)&Bs[(nt * 16 + mrow) * 32 + khalf * 16];
            c1[nt] = __builtin_amdgcn_wmma_f32_16x16x32_f16(
                false, a.v, false, b, (short)0, c1[nt], false, false);
        }
    }

    #pragma unroll
    for (int nt = 0; nt < 8; ++nt) {
        int col = nt * 16 + mrow;
        float bias = b1[col];
        #pragma unroll
        for (int r = 0; r < 8; ++r) {
            int row = wave * 16 + r + khalf * 8;
            float v = c1[nt][r] + bias;
            Hs[row * 128 + col] = (_Float16)(v > 0.0f ? v : 0.0f);
        }
    }

    v8f c2[8];
    #pragma unroll
    for (int i = 0; i < 8; ++i) c2[i] = {};
    #pragma unroll
    for (int kc = 0; kc < 4; ++kc) {
        int kb = kc * 32;
        AFrag a;
        a.h[0] = *(const v8h*)&Hs[(wave * 16 + mrow) * 128 + kb + khalf * 16];
        a.h[1] = *(const v8h*)&Hs[(wave * 16 + mrow) * 128 + kb + khalf * 16 + 8];
        #pragma unroll
        for (int nt = 0; nt < 8; ++nt) {
            v16h b = *(const v16h*)&W2s[(nt * 16 + mrow) * 128 + kb + khalf * 16];
            c2[nt] = __builtin_amdgcn_wmma_f32_16x16x32_f16(
                false, a.v, false, b, (short)0, c2[nt], false, false);
        }
    }

    // residual update
    #pragma unroll
    for (int nt = 0; nt < 8; ++nt) {
        int col = nt * 16 + mrow;
        float bias = b2[col];
        #pragma unroll
        for (int r = 0; r < 8; ++r) {
            int row = n0 + wave * 16 + r + khalf * 8;
            if (row < N) {
                float* xp = x + (size_t)row * HIDDEN + col;
                *xp = *xp + c2[nt][r] + bias;
            }
        }
    }
}

// decode + row-normalize
__global__ void decoder_kernel(const float* __restrict__ x, const float* __restrict__ dec_w,
                               const float* __restrict__ dec_b, float* __restrict__ out, int N) {
    int n = blockIdx.x * blockDim.x + threadIdx.x;
    if (n >= N) return;
    float o0 = dec_b[0], o1 = dec_b[1], o2 = dec_b[2];
    const float* xr = x + (size_t)n * HIDDEN;
    #pragma unroll 4
    for (int k = 0; k < HIDDEN; ++k) {
        float xv = xr[k];
        o0 += xv * dec_w[3 * k + 0];
        o1 += xv * dec_w[3 * k + 1];
        o2 += xv * dec_w[3 * k + 2];
    }
    float nrm = fmaxf(sqrtf(o0 * o0 + o1 * o1 + o2 * o2), 1e-12f);
    out[3 * n + 0] = o0 / nrm;
    out[3 * n + 1] = o1 / nrm;
    out[3 * n + 2] = o2 / nrm;
}

// ---------------------------------------------------------------------------
extern "C" void kernel_launch(void* const* d_in, const int* in_sizes, int n_in,
                              void* d_out, int out_size, void* d_ws, size_t ws_size,
                              hipStream_t stream) {
    const float* edge_attr = (const float*)d_in[1];
    const int*   edge_index = (const int*)d_in[2];
    const float* enc_w = (const float*)d_in[3];
    const float* enc_b = (const float*)d_in[4];
    const float* dec_w = (const float*)d_in[5];
    const float* dec_b = (const float*)d_in[6];
    const float* edge_w1 = (const float*)d_in[7];
    const float* edge_b1 = (const float*)d_in[8];
    const float* edge_w2 = (const float*)d_in[9];
    const float* edge_b2 = (const float*)d_in[10];
    const float* node_w1 = (const float*)d_in[11];
    const float* node_b1 = (const float*)d_in[12];
    const float* node_w2 = (const float*)d_in[13];
    const float* node_b2 = (const float*)d_in[14];

    const int N = in_sizes[0] / 3;   // nodes (from pos)
    const int E = in_sizes[1] / 3;   // edges (from edge_attr)

    // ----- workspace carve -----
    char* ws = (char*)d_ws;
    size_t off = 0;
    auto carve = [&](size_t bytes) {
        void* p = ws + off;
        off = (off + bytes + 255) & ~(size_t)255;
        return p;
    };
    float*     x    = (float*)carve((size_t)N * HIDDEN * 4);
    float*     sums = (float*)carve((size_t)N * HIDDEN * 4);
    float*     cnt  = (float*)carve((size_t)N * 4);
    _Float16*  w1t  = (_Float16*)carve((size_t)NLAYERS * 128 * 384 * 2);
    _Float16*  w2t  = (_Float16*)carve((size_t)NLAYERS * 128 * 128 * 2);
    _Float16*  nw1t = (_Float16*)carve((size_t)NLAYERS * 128 * 256 * 2);
    _Float16*  nw2t = (_Float16*)carve((size_t)NLAYERS * 128 * 128 * 2);
    _Float16*  ef   = (_Float16*)carve((size_t)E * HIDDEN * 2);   // f16 edge state
    (void)ws_size;

    // ----- weight convert+transpose to f16 [n][k] -----
    {
        int t1 = NLAYERS * 384 * 128;
        convt_kernel<<<(t1 + 255) / 256, 256, 0, stream>>>(edge_w1, w1t, 384, 128, t1);
        int t2 = NLAYERS * 128 * 128;
        convt_kernel<<<(t2 + 255) / 256, 256, 0, stream>>>(edge_w2, w2t, 128, 128, t2);
        int t3 = NLAYERS * 256 * 128;
        convt_kernel<<<(t3 + 255) / 256, 256, 0, stream>>>(node_w1, nw1t, 256, 128, t3);
        int t4 = NLAYERS * 128 * 128;
        convt_kernel<<<(t4 + 255) / 256, 256, 0, stream>>>(node_w2, nw2t, 128, 128, t4);
    }

    // ----- init: x = 0, cnt = degree(dst), ef = encoder(edge_attr) -----
    {
        size_t nx = (size_t)N * HIDDEN;
        zero_kernel<<<(unsigned)((nx + 255) / 256), 256, 0, stream>>>(x, nx);
        zero_kernel<<<(unsigned)((N + 255) / 256), 256, 0, stream>>>(cnt, (size_t)N);
        count_kernel<<<(E + 255) / 256, 256, 0, stream>>>(edge_index, cnt, E);
        size_t ne = (size_t)E * HIDDEN;
        encoder_kernel<<<(unsigned)((ne + 255) / 256), 256, 0, stream>>>(edge_attr, enc_w, enc_b, ef, E);
    }

    const int edge_blocks = (E + 63) / 64;
    const int node_blocks = (N + 63) / 64;
    size_t nsum = (size_t)N * HIDDEN;

    for (int l = 0; l < NLAYERS; ++l) {
        zero_kernel<<<(unsigned)((nsum + 255) / 256), 256, 0, stream>>>(sums, nsum);
        edge_mlp_kernel<<<edge_blocks, 128, 0, stream>>>(
            x, ef, sums, edge_index,
            w1t + (size_t)l * 128 * 384, w2t + (size_t)l * 128 * 128,
            edge_b1 + l * HIDDEN, edge_b2 + l * HIDDEN, E);
        node_mlp_kernel<<<node_blocks, 128, 0, stream>>>(
            x, sums, cnt,
            nw1t + (size_t)l * 128 * 256, nw2t + (size_t)l * 128 * 128,
            node_b1 + l * HIDDEN, node_b2 + l * HIDDEN, N);
    }

    decoder_kernel<<<(N + 255) / 256, 256, 0, stream>>>(x, dec_w, dec_b, (float*)d_out, N);
}
